// VQVAE_5763846111562
// MI455X (gfx1250) — matrix operand order repaired and checked by reading, not compiled, not run on hardware
//
#include <hip/hip_runtime.h>
#include <hip/hip_bf16.h>
#include <math.h>

// ---------------------------------------------------------------------------
// CDNA5 (gfx1250) VQVAE forward. wave32. All GEMM-like ops (conv, conv-
// transpose, attention projections) run on v_wmma_f32_16x16x32_f16 with a
// 128x64 block tile (8 waves x 16M x 64N), cooperative shared B staging,
// double-buffered LDS software pipeline, branch-free predicated loads.
// ---------------------------------------------------------------------------

typedef __attribute__((ext_vector_type(16))) _Float16 v16h;
typedef __attribute__((ext_vector_type(4)))  _Float16 v4h;
typedef __attribute__((ext_vector_type(8)))  float    v8f;

#define TILE_K 32
#define WPB 8            // waves per block (256 threads, wave32)

__device__ __forceinline__ float silu_f(float x) { return x / (1.0f + __expf(-x)); }

// ---- WMMA fragment assembly in contiguous k = 16*(lane/16) + s order (16-bit A 16x32 / B 32x16) -----
__device__ __forceinline__ v16h load_a_frag(const _Float16* As /*[16][32] row-major*/) {
    const int lane = threadIdx.x & 31;
    const int m  = lane & 15;
    const int kb = (lane & 16) ? 16 : 0;
    v16h a;
#pragma unroll
    for (int i = 0; i < 8; ++i) a[i]     = As[m * TILE_K + kb + i];
#pragma unroll
    for (int i = 0; i < 8; ++i) a[8 + i] = As[m * TILE_K + kb + 8 + i];
    return a;
}
__device__ __forceinline__ v16h load_b_frag(const _Float16* Bs /*[16 n][32 k] n-major*/) {
    const int lane = threadIdx.x & 31;
    const int n  = lane & 15;
    const int kb = (lane & 16) ? 16 : 0;
    v16h b;
#pragma unroll
    for (int i = 0; i < 16; ++i) b[i] = Bs[n * TILE_K + kb + i];
    return b;
}
// pack 16 floats -> 4x ds_store_b64 at a 16-half-aligned LDS slot
__device__ __forceinline__ void commit16(_Float16* dst16, const float* v) {
    v4h* dst = (v4h*)dst16;
#pragma unroll
    for (int j = 0; j < 4; ++j) {
        v4h h;
        h[0] = (_Float16)v[4 * j + 0]; h[1] = (_Float16)v[4 * j + 1];
        h[2] = (_Float16)v[4 * j + 2]; h[3] = (_Float16)v[4 * j + 3];
        dst[j] = h;
    }
}
__device__ __forceinline__ void commit8(_Float16* dst16, const float* v) {
    v4h* dst = (v4h*)dst16;
#pragma unroll
    for (int j = 0; j < 2; ++j) {
        v4h h;
        h[0] = (_Float16)v[4 * j + 0]; h[1] = (_Float16)v[4 * j + 1];
        h[2] = (_Float16)v[4 * j + 2]; h[3] = (_Float16)v[4 * j + 3];
        dst[j] = h;
    }
}

// ---------------------------------------------------------------------------
// Implicit-GEMM conv2d, compile-time (KH,KW,STRIDE). Double-buffered pipeline.
// fuse: 0 = bias only, 1 = bias + SiLU.
// ---------------------------------------------------------------------------
template <int KH, int KW, int STRIDE>
__global__ __launch_bounds__(256)
void k_conv2d_t(const float* __restrict__ x, const float* __restrict__ w,
                const float* __restrict__ bias, float* __restrict__ y,
                int N, int Cin, int H, int W, int Cout,
                int pad, int OH, int OW, int fuse)
{
    constexpr int KHW = KH * KW;
    __shared__ __align__(16) _Float16 smA[2][WPB][16 * TILE_K];
    __shared__ __align__(16) _Float16 smB[2][4][16 * TILE_K];
    const int wave = threadIdx.x >> 5;
    const int lane = threadIdx.x & 31;
    const int OHW = OH * OW;
    const int M = N * OHW;
    const int K = Cin * KHW;
    const int mbase = (blockIdx.x * WPB + wave) * 16;
    const int nbase = blockIdx.y * 64;

    // per-lane A staging coords (decoded once)
    const int arow  = lane & 15;
    const int ahalf = (lane >> 4) & 1;
    const int am    = mbase + arow;
    const bool am_ok = (am < M);
    int an = 0, aoy = 0, aox = 0;
    if (am_ok) { an = am / OHW; const int r = am - an * OHW; aoy = r / OW; aox = r - aoy * OW; }
    const long xbase = (long)an * Cin * H * W;

    // per-thread cooperative B staging coords
    const int nl = threadIdx.x >> 2;          // 0..63
    const int ks = (threadIdx.x & 3) * 8;     // 0,8,16,24
    const int bn = nbase + nl;
    const int bsub = nl >> 4, bnr = nl & 15;

    // branch-free predicated tile loads into registers
    auto loadA = [&](int kt, float* ar) {
#pragma unroll
        for (int i = 0; i < 16; ++i) {
            const int k = kt + ahalf * 16 + i;
            const int ci = k / KHW; const int r = k - ci * KHW;
            const int kh = r / KW;  const int kw = r - kh * KW;
            const int iy = aoy * STRIDE - pad + kh;
            const int ix = aox * STRIDE - pad + kw;
            const bool val = am_ok && k < K && iy >= 0 && iy < H && ix >= 0 && ix < W;
            const long off = val ? (xbase + ((long)ci * H + iy) * W + ix) : 0;
            const float v = x[off];
            ar[i] = val ? v : 0.f;
        }
    };
    auto loadB = [&](int kt, float* br) {
#pragma unroll
        for (int i = 0; i < 8; ++i) {
            const int k = kt + ks + i;
            const bool val = bn < Cout && k < K;
            const long off = val ? ((long)bn * K + k) : 0;
            const float v = w[off];
            br[i] = val ? v : 0.f;
        }
    };

    const v8f vzero = {0.f, 0.f, 0.f, 0.f, 0.f, 0.f, 0.f, 0.f};
    v8f acc[4] = {vzero, vzero, vzero, vzero};

    // prologue: stage tile 0 into buffer 0
    {
        float ar[16], br[8];
        loadA(0, ar); loadB(0, br);
        commit16(&smA[0][wave][arow * TILE_K + ahalf * 16], ar);
        commit8(&smB[0][bsub][bnr * TILE_K + ks], br);
    }
    __syncthreads();

    int buf = 0;
    for (int k0 = 0; k0 < K; k0 += TILE_K) {
        const int nxt = buf ^ 1;
        const bool more = (k0 + TILE_K) < K;
        float ar[16], br[8];
        if (more) { loadA(k0 + TILE_K, ar); loadB(k0 + TILE_K, br); }   // overlap w/ WMMA
        const v16h a = load_a_frag(smA[buf][wave]);
#pragma unroll
        for (int j = 0; j < 4; ++j)
            acc[j] = __builtin_amdgcn_wmma_f32_16x16x32_f16(
                false, a, false, load_b_frag(smB[buf][j]), (short)0, acc[j], false, false);
        if (more) {
            commit16(&smA[nxt][wave][arow * TILE_K + ahalf * 16], ar);
            commit8(&smB[nxt][bsub][bnr * TILE_K + ks], br);
        }
        __syncthreads();
        buf = nxt;
    }

    // epilogue: incremental (n,oy,ox) decode, 4 channel sub-tiles
    const int lane15 = lane & 15;
    const int rofs = (lane & 16) ? 8 : 0;
    const int m0 = mbase + rofs;
    int n_i = 0, oy = 0, ox = 0;
    if (m0 < M) { n_i = m0 / OHW; const int rem = m0 - n_i * OHW; oy = rem / OW; ox = rem - oy * OW; }
    int ocv[4]; float bv[4];
#pragma unroll
    for (int j = 0; j < 4; ++j) {
        ocv[j] = nbase + j * 16 + lane15;
        bv[j] = (ocv[j] < Cout) ? bias[ocv[j]] : 0.f;
    }
#pragma unroll
    for (int r = 0; r < 8; ++r) {
        const int m = m0 + r;
        if (m < M) {
            const long basep = (long)n_i * Cout * OHW + (oy * OW + ox);
#pragma unroll
            for (int j = 0; j < 4; ++j) {
                if (ocv[j] < Cout) {
                    float v = acc[j][r] + bv[j];
                    if (fuse) v = silu_f(v);
                    y[basep + (long)ocv[j] * OHW] = v;
                }
            }
            ++ox;
            if (ox == OW) { ox = 0; ++oy; if (oy == OH) { oy = 0; ++n_i; } }
        }
    }
}

// ---------------------------------------------------------------------------
// ConvTranspose2d(k=4, s=2, p=1) as lhs-dilated implicit GEMM (flipped
// kernel). weight stored (in=C, out=C, 4, 4). K = C*16. Same pipeline.
// ---------------------------------------------------------------------------
__global__ __launch_bounds__(256)
void k_convt2d(const float* __restrict__ x, const float* __restrict__ w,
               const float* __restrict__ bias, float* __restrict__ y,
               int N, int C, int IH, int IW, int OH, int OW)
{
    __shared__ __align__(16) _Float16 smA[2][WPB][16 * TILE_K];
    __shared__ __align__(16) _Float16 smB[2][4][16 * TILE_K];
    const int wave = threadIdx.x >> 5;
    const int lane = threadIdx.x & 31;
    const int OHW = OH * OW;
    const int M = N * OHW;
    const int K = C * 16;
    const int mbase = (blockIdx.x * WPB + wave) * 16;
    const int nbase = blockIdx.y * 64;

    const int arow  = lane & 15;
    const int ahalf = (lane >> 4) & 1;
    const int am    = mbase + arow;
    const bool am_ok = (am < M);
    int an = 0, aoy = 0, aox = 0;
    if (am_ok) { an = am / OHW; const int r = am - an * OHW; aoy = r / OW; aox = r - aoy * OW; }
    const long xbase = (long)an * C * IH * IW;

    const int nl = threadIdx.x >> 2;
    const int ks = (threadIdx.x & 3) * 8;
    const int bn = nbase + nl;
    const int bsub = nl >> 4, bnr = nl & 15;

    auto loadA = [&](int kt, float* ar) {
#pragma unroll
        for (int i = 0; i < 16; ++i) {
            const int k = kt + ahalf * 16 + i;
            const int ci = k >> 4; const int r = k & 15;
            const int kh = r >> 2; const int kw = r & 3;
            const int ly = aoy - 2 + kh;
            const int lx = aox - 2 + kw;
            const int iy = ly >> 1, ix = lx >> 1;
            const bool val = am_ok && ly >= 0 && lx >= 0 && !(ly & 1) && !(lx & 1) &&
                             iy < IH && ix < IW;
            const long off = val ? (xbase + ((long)ci * IH + iy) * IW + ix) : 0;
            const float v = x[off];
            ar[i] = val ? v : 0.f;
        }
    };
    auto loadB = [&](int kt, float* br) {
#pragma unroll
        for (int i = 0; i < 8; ++i) {
            const int k = kt + ks + i;
            const int ci = k >> 4; const int r = k & 15;
            const int kh = r >> 2; const int kw = r & 3;
            const bool val = bn < C;
            const long off = val ? (((long)ci * C + bn) * 16 + (3 - kh) * 4 + (3 - kw)) : 0;
            const float v = w[off];
            br[i] = val ? v : 0.f;
        }
    };

    const v8f vzero = {0.f, 0.f, 0.f, 0.f, 0.f, 0.f, 0.f, 0.f};
    v8f acc[4] = {vzero, vzero, vzero, vzero};

    {
        float ar[16], br[8];
        loadA(0, ar); loadB(0, br);
        commit16(&smA[0][wave][arow * TILE_K + ahalf * 16], ar);
        commit8(&smB[0][bsub][bnr * TILE_K + ks], br);
    }
    __syncthreads();

    int buf = 0;
    for (int k0 = 0; k0 < K; k0 += TILE_K) {
        const int nxt = buf ^ 1;
        const bool more = (k0 + TILE_K) < K;
        float ar[16], br[8];
        if (more) { loadA(k0 + TILE_K, ar); loadB(k0 + TILE_K, br); }
        const v16h a = load_a_frag(smA[buf][wave]);
#pragma unroll
        for (int j = 0; j < 4; ++j)
            acc[j] = __builtin_amdgcn_wmma_f32_16x16x32_f16(
                false, a, false, load_b_frag(smB[buf][j]), (short)0, acc[j], false, false);
        if (more) {
            commit16(&smA[nxt][wave][arow * TILE_K + ahalf * 16], ar);
            commit8(&smB[nxt][bsub][bnr * TILE_K + ks], br);
        }
        __syncthreads();
        buf = nxt;
    }

    const int lane15 = lane & 15;
    const int rofs = (lane & 16) ? 8 : 0;
    const int m0 = mbase + rofs;
    int n_i = 0, oy = 0, ox = 0;
    if (m0 < M) { n_i = m0 / OHW; const int rem = m0 - n_i * OHW; oy = rem / OW; ox = rem - oy * OW; }
    int ocv[4]; float bv[4];
#pragma unroll
    for (int j = 0; j < 4; ++j) {
        ocv[j] = nbase + j * 16 + lane15;
        bv[j] = (ocv[j] < C) ? bias[ocv[j]] : 0.f;
    }
#pragma unroll
    for (int r = 0; r < 8; ++r) {
        const int m = m0 + r;
        if (m < M) {
            const long basep = (long)n_i * C * OHW + (oy * OW + ox);
#pragma unroll
            for (int j = 0; j < 4; ++j)
                if (ocv[j] < C) y[basep + (long)ocv[j] * OHW] = acc[j][r] + bv[j];
            ++ox;
            if (ox == OW) { ox = 0; ++oy; if (oy == OH) { oy = 0; ++n_i; } }
        }
    }
}

// ---------------------------------------------------------------------------
// Row-major GEMM: C[M,N] = A[M,K] @ B[N,K]^T + bias (torch Linear semantics)
// ---------------------------------------------------------------------------
__global__ __launch_bounds__(256)
void k_gemm_nt(const float* __restrict__ A, const float* __restrict__ B,
               const float* __restrict__ bias, float* __restrict__ Cm,
               int M, int Nn, int K)
{
    __shared__ __align__(16) _Float16 smA[2][WPB][16 * TILE_K];
    __shared__ __align__(16) _Float16 smB[2][4][16 * TILE_K];
    const int wave = threadIdx.x >> 5;
    const int lane = threadIdx.x & 31;
    const int mbase = (blockIdx.x * WPB + wave) * 16;
    const int nbase = blockIdx.y * 64;
    const int arow  = lane & 15;
    const int ahalf = (lane >> 4) & 1;
    const int am = mbase + arow;

    const int nl = threadIdx.x >> 2;
    const int ks = (threadIdx.x & 3) * 8;
    const int bn = nbase + nl;
    const int bsub = nl >> 4, bnr = nl & 15;

    auto loadA = [&](int kt, float* ar) {
#pragma unroll
        for (int i = 0; i < 16; ++i) {
            const int k = kt + ahalf * 16 + i;
            const bool val = (am < M) && (k < K);
            const long off = val ? ((long)am * K + k) : 0;
            const float v = A[off];
            ar[i] = val ? v : 0.f;
        }
    };
    auto loadB = [&](int kt, float* br) {
#pragma unroll
        for (int i = 0; i < 8; ++i) {
            const int k = kt + ks + i;
            const bool val = (bn < Nn) && (k < K);
            const long off = val ? ((long)bn * K + k) : 0;
            const float v = B[off];
            br[i] = val ? v : 0.f;
        }
    };

    const v8f vzero = {0.f, 0.f, 0.f, 0.f, 0.f, 0.f, 0.f, 0.f};
    v8f acc[4] = {vzero, vzero, vzero, vzero};

    {
        float ar[16], br[8];
        loadA(0, ar); loadB(0, br);
        commit16(&smA[0][wave][arow * TILE_K + ahalf * 16], ar);
        commit8(&smB[0][bsub][bnr * TILE_K + ks], br);
    }
    __syncthreads();

    int buf = 0;
    for (int k0 = 0; k0 < K; k0 += TILE_K) {
        const int nxt = buf ^ 1;
        const bool more = (k0 + TILE_K) < K;
        float ar[16], br[8];
        if (more) { loadA(k0 + TILE_K, ar); loadB(k0 + TILE_K, br); }
        const v16h a = load_a_frag(smA[buf][wave]);
#pragma unroll
        for (int j = 0; j < 4; ++j)
            acc[j] = __builtin_amdgcn_wmma_f32_16x16x32_f16(
                false, a, false, load_b_frag(smB[buf][j]), (short)0, acc[j], false, false);
        if (more) {
            commit16(&smA[nxt][wave][arow * TILE_K + ahalf * 16], ar);
            commit8(&smB[nxt][bsub][bnr * TILE_K + ks], br);
        }
        __syncthreads();
        buf = nxt;
    }

    const int lane15 = lane & 15;
    const int rofs = (lane & 16) ? 8 : 0;
#pragma unroll
    for (int j = 0; j < 4; ++j) {
        const int n = nbase + j * 16 + lane15;
        if (n < Nn) {
            const float bv = bias ? bias[n] : 0.f;
#pragma unroll
            for (int r = 0; r < 8; ++r) {
                const int m = mbase + r + rofs;
                if (m < M) Cm[(long)m * Nn + n] = acc[j][r] + bv;
            }
        }
    }
}

// ---------------------------------------------------------------------------
// GroupNorm (G=32) + optional fused SiLU. One block per (n, group).
// float4 passes (all HW sizes divisible by 4), FMA normalize.
// ---------------------------------------------------------------------------
__global__ __launch_bounds__(256)
void k_groupnorm(const float* __restrict__ x, const float* __restrict__ gamma,
                 const float* __restrict__ beta, float* __restrict__ y,
                 int C, int HW, int Gn, int do_silu)
{
    const int bid = blockIdx.x;
    const int n = bid / Gn, grp = bid - n * Gn;
    const int cpg = C / Gn;
    const long cnt = (long)cpg * HW;
    const long base = ((long)n * C + (long)grp * cpg) * HW;
    const float4* x4 = (const float4*)(x + base);
    const long cnt4 = cnt >> 2;

    float s = 0.f, ss = 0.f;
    for (long i = threadIdx.x; i < cnt4; i += blockDim.x) {
        const float4 v = x4[i];
        s += v.x + v.y + v.z + v.w;
        ss += v.x * v.x + v.y * v.y + v.z * v.z + v.w * v.w;
    }
#pragma unroll
    for (int o = 16; o > 0; o >>= 1) { s += __shfl_xor(s, o, 32); ss += __shfl_xor(ss, o, 32); }
    __shared__ float red[16];
    const int wid = threadIdx.x >> 5, lane = threadIdx.x & 31;
    if (lane == 0) { red[wid] = s; red[8 + wid] = ss; }
    __syncthreads();
    if (threadIdx.x == 0) {
        float ts = 0.f, tss = 0.f;
        for (int i = 0; i < 8; ++i) { ts += red[i]; tss += red[8 + i]; }
        red[0] = ts; red[8] = tss;
    }
    __syncthreads();
    const float mu = red[0] / (float)cnt;
    const float var = red[8] / (float)cnt - mu * mu;
    const float rstd = rsqrtf(var + 1e-5f);

    const int HW4 = HW >> 2;
    for (int cl = 0; cl < cpg; ++cl) {
        const int c = grp * cpg + cl;
        const float gsc = gamma[c] * rstd;
        const float bof = beta[c] - mu * gsc;
        const float4* xc = (const float4*)(x + base + (long)cl * HW);
        float4* yc = (float4*)(y + base + (long)cl * HW);
        for (int j = threadIdx.x; j < HW4; j += blockDim.x) {
            float4 v = xc[j], o;
            o.x = fmaf(v.x, gsc, bof); o.y = fmaf(v.y, gsc, bof);
            o.z = fmaf(v.z, gsc, bof); o.w = fmaf(v.w, gsc, bof);
            if (do_silu) { o.x = silu_f(o.x); o.y = silu_f(o.y); o.z = silu_f(o.z); o.w = silu_f(o.w); }
            yc[j] = o;
        }
    }
}

// y = (a + b), optional SiLU; float4 vectorized (all sizes are /4)
__global__ void k_add_act(const float4* __restrict__ a, const float4* __restrict__ b,
                          float4* __restrict__ y, long n4, int do_silu)
{
    for (long i = (long)blockIdx.x * blockDim.x + threadIdx.x; i < n4;
         i += (long)gridDim.x * blockDim.x) {
        float4 va = a[i], vb = b[i], vo;
        vo.x = va.x + vb.x; vo.y = va.y + vb.y; vo.z = va.z + vb.z; vo.w = va.w + vb.w;
        if (do_silu) { vo.x = silu_f(vo.x); vo.y = silu_f(vo.y); vo.z = silu_f(vo.z); vo.w = silu_f(vo.w); }
        y[i] = vo;
    }
}

// ---------------------------------------------------------------------------
// Batched tiled transpose: per batch, in (R x Cc) row-major -> out (Cc x R).
// grid = (ceil(Cc/32), ceil(R/32), B), block = (32, 8). Both sides coalesced.
// ---------------------------------------------------------------------------
__global__ __launch_bounds__(256)
void k_tr(const float* __restrict__ in, float* __restrict__ out, int R, int Cc)
{
    __shared__ float tile[32][33];
    const long bofs = (long)blockIdx.z * R * Cc;
    const float* ib = in + bofs;
    float* ob = out + bofs;
    const int r0 = blockIdx.y * 32, c0 = blockIdx.x * 32;
#pragma unroll
    for (int i = threadIdx.y; i < 32; i += 8) {
        const int r = r0 + i, c = c0 + threadIdx.x;
        if (r < R && c < Cc) tile[i][threadIdx.x] = ib[(long)r * Cc + c];
    }
    __syncthreads();
#pragma unroll
    for (int i = threadIdx.y; i < 32; i += 8) {
        const int c = c0 + i, r = r0 + threadIdx.x;
        if (c < Cc && r < R) ob[(long)c * R + r] = tile[threadIdx.x][i];
    }
}

// ---------------------------------------------------------------------------
// Attention core: L=8 queries, N=1024 batch, heads=2, hd=128. Block per (n,h).
// ---------------------------------------------------------------------------
__global__ __launch_bounds__(128)
void k_attn(const float* __restrict__ qkv, float* __restrict__ out, int N, int E)
{
    const int L = 8;
    const int hd = E / 2;                   // 128 == blockDim.x
    __shared__ float qs[8][128], ks[8][128], vs[8][128], sc[8][8];
    const int n = blockIdx.x, h = blockIdx.y;
    const int t = threadIdx.x;

    for (int l = 0; l < L; ++l) {
        const long row = (long)(l * N + n) * (3 * E);
        qs[l][t] = qkv[row + h * hd + t];
        ks[l][t] = qkv[row + E + h * hd + t];
        vs[l][t] = qkv[row + 2 * E + h * hd + t];
    }
    __syncthreads();
    if (t < L * L) {
        const int i = t >> 3, j = t & 7;
        float s = 0.f;
        for (int d = 0; d < hd; ++d) s += qs[i][d] * ks[j][d];
        sc[i][j] = s * rsqrtf((float)hd);
    }
    __syncthreads();
    if (t < L) {
        float mx = -1e30f;
        for (int j = 0; j < L; ++j) mx = fmaxf(mx, sc[t][j]);
        float sum = 0.f;
        for (int j = 0; j < L; ++j) { const float e = __expf(sc[t][j] - mx); sc[t][j] = e; sum += e; }
        const float inv = 1.f / sum;
        for (int j = 0; j < L; ++j) sc[t][j] *= inv;
    }
    __syncthreads();
    for (int i = 0; i < L; ++i) {
        float o = 0.f;
        for (int j = 0; j < L; ++j) o += sc[i][j] * vs[j][t];
        out[(long)(i * N + n) * E + h * hd + t] = o;
    }
}

// ---------------------------------------------------------------------------
// VQ: per-row argmin over 8192x3 codebook (LDS-tiled), straight-through q,
// and sum of (q-x)^2 for the (identical) cb/commit losses.
// ---------------------------------------------------------------------------
#define VQ_TILE 1024
__global__ __launch_bounds__(256)
void k_vq(const float* __restrict__ xf, const float* __restrict__ emb,
          float* __restrict__ qrows, float* __restrict__ sum_out, int M, int CB)
{
    __shared__ float se[VQ_TILE * 3];
    const int m = blockIdx.x * blockDim.x + threadIdx.x;
    const bool ok = (m < M);
    float x0 = 0.f, x1 = 0.f, x2 = 0.f;
    if (ok) { x0 = xf[m * 3]; x1 = xf[m * 3 + 1]; x2 = xf[m * 3 + 2]; }
    float best = 3.4e38f; int bi = 0;
    for (int t0 = 0; t0 < CB; t0 += VQ_TILE) {
        for (int i = threadIdx.x; i < VQ_TILE * 3; i += blockDim.x) se[i] = emb[t0 * 3 + i];
        __syncthreads();
        for (int j = 0; j < VQ_TILE; ++j) {
            const float d0 = x0 - se[j * 3], d1 = x1 - se[j * 3 + 1], d2 = x2 - se[j * 3 + 2];
            const float d = d0 * d0 + d1 * d1 + d2 * d2;
            if (d < best) { best = d; bi = t0 + j; }
        }
        __syncthreads();
    }
    float lsum = 0.f;
    if (ok) {
        const float q0 = emb[bi * 3], q1 = emb[bi * 3 + 1], q2 = emb[bi * 3 + 2];
        qrows[m * 3] = q0; qrows[m * 3 + 1] = q1; qrows[m * 3 + 2] = q2;
        const float d0 = q0 - x0, d1 = q1 - x1, d2 = q2 - x2;
        lsum = d0 * d0 + d1 * d1 + d2 * d2;
    }
#pragma unroll
    for (int o = 16; o > 0; o >>= 1) lsum += __shfl_xor(lsum, o, 32);
    __shared__ float red[8];
    if ((threadIdx.x & 31) == 0) red[threadIdx.x >> 5] = lsum;
    __syncthreads();
    if (threadIdx.x == 0) {
        float s = 0.f;
        for (int i = 0; i < 8; ++i) s += red[i];
        atomicAdd(sum_out, s);
    }
}

__global__ void k_zero(float* p, int n) { for (int i = threadIdx.x; i < n; i += blockDim.x) p[i] = 0.f; }
__global__ void k_fin(const float* __restrict__ sum, float* __restrict__ dst, float inv)
{ if (threadIdx.x == 0) { dst[0] = sum[0] * inv; dst[1] = sum[0] * inv; } }

// ===========================================================================
// Host orchestration
// ===========================================================================
static inline int cdiv_i(int a, int b) { return (a + b - 1) / b; }

extern "C" void kernel_launch(void* const* d_in, const int* in_sizes, int n_in,
                              void* d_out, int out_size, void* d_ws, size_t ws_size,
                              hipStream_t stream)
{
    (void)in_sizes; (void)n_in; (void)out_size; (void)ws_size;
    auto IN = [&](int i) { return (const float*)d_in[i]; };
    float* OUT = (float*)d_out;
    float* ws  = (float*)d_ws;

    // d_out layout: d[1572864] | z[24576] | cb | commit
    const long ZOFF = 1572864, SOFF = ZOFF + 24576;

    // workspace carve: 3 big ping-pong buffers + small slots
    const long BIGN = 67108864;   // 8*128*256*256 floats (268MB)
    const long SMN  = 8388608;    // 32MB
    float* B0 = ws;
    float* B1 = B0 + BIGN;
    float* B2 = B1 + BIGN;
    float* S0 = B2 + BIGN;        // attention qkv / vq rows in
    float* S1 = S0 + SMN;         // attention core out / vq q rows
    float* B3 = S1 + SMN;         // mid-block input slot (8MB)
    float* RED = B3 + 2097152;

    // ---- layer wrappers ---------------------------------------------------
    auto conv = [&](const float* x, const float* w, const float* b, float* y,
                    int Cin, int H, int Cout, int KH, int st, int pad, int fuse) {
        const int OH = (H + 2 * pad - KH) / st + 1;
        const int M  = 8 * OH * OH;
        dim3 g(cdiv_i(M, 128), cdiv_i(Cout, 64));
        if (KH == 3)
            k_conv2d_t<3, 3, 1><<<g, 256, 0, stream>>>(x, w, b, y, 8, Cin, H, H, Cout, pad, OH, OH, fuse);
        else if (KH == 1)
            k_conv2d_t<1, 1, 1><<<g, 256, 0, stream>>>(x, w, b, y, 8, Cin, H, H, Cout, pad, OH, OH, fuse);
        else
            k_conv2d_t<4, 4, 2><<<g, 256, 0, stream>>>(x, w, b, y, 8, Cin, H, H, Cout, pad, OH, OH, fuse);
    };
    auto convt = [&](const float* x, const float* w, const float* b, float* y, int C, int H) {
        const int OH = 2 * H, M = 8 * OH * OH;
        dim3 g(cdiv_i(M, 128), cdiv_i(C, 64));
        k_convt2d<<<g, 256, 0, stream>>>(x, w, b, y, 8, C, H, H, OH, OH);
    };
    auto gemm = [&](const float* A, const float* Bm, const float* bias, float* Cm,
                    int M, int Nn, int K) {
        dim3 g(cdiv_i(M, 128), cdiv_i(Nn, 64));
        k_gemm_nt<<<g, 256, 0, stream>>>(A, Bm, bias, Cm, M, Nn, K);
    };
    auto gn = [&](const float* x, const float* g_, const float* b_, float* y,
                  int C, int HW, int dosilu) {
        k_groupnorm<<<8 * 32, 256, 0, stream>>>(x, g_, b_, y, C, HW, 32, dosilu);
    };
    auto add = [&](const float* a, const float* b, float* y, long n, int dosilu) {
        const long n4 = n / 4;
        int blocks = (int)((n4 + 255) / 256); if (blocks > 65535) blocks = 65535;
        k_add_act<<<blocks, 256, 0, stream>>>((const float4*)a, (const float4*)b, (float4*)y, n4, dosilu);
    };
    // (B,C,P) row-major -> rows (B*P, C):   per-batch transpose of (C x P)
    auto t_fwd = [&](const float* in, float* out, int C, int P) {
        dim3 g(cdiv_i(P, 32), cdiv_i(C, 32), 8);
        k_tr<<<g, dim3(32, 8), 0, stream>>>(in, out, C, P);
    };
    // rows (B*P, C) -> (B,C,P):             per-batch transpose of (P x C)
    auto t_inv = [&](const float* in, float* out, int C, int P) {
        dim3 g(cdiv_i(C, 32), cdiv_i(P, 32), 8);
        k_tr<<<g, dim3(32, 8), 0, stream>>>(in, out, P, C);
    };

    // ---- block helpers (leaf order = jax sorted-key pytree flatten) -------
    // down: +0 c1.b +1 c1.w +2 c2.b +3 c2.w +4 down.b +5 down.w
    //       +6 gn1.b +7 gn1.g +8 gn2.b +9 gn2.g +10 res.b +11 res.w
    auto down_block = [&](int base, int ci, int co, int H, float* yout) {
        conv(B0, IN(base + 11), IN(base + 10), B1, ci, H, co, 1, 1, 0, 0);     // res
        gn(B0, IN(base + 7), IN(base + 6), B2, ci, H * H, 1);                  // gn1+silu
        conv(B2, IN(base + 1), IN(base + 0), B0, ci, H, co, 3, 1, 1, 0);       // c1
        gn(B0, IN(base + 9), IN(base + 8), B2, co, H * H, 1);                  // gn2+silu
        conv(B2, IN(base + 3), IN(base + 2), B0, co, H, co, 3, 1, 1, 0);       // c2
        add(B0, B1, B2, (long)8 * co * H * H, 1);                              // silu(x+res)
        conv(B2, IN(base + 5), IN(base + 4), yout, co, H, co, 4, 2, 1, 0);     // down 4x4 s2
    };
    // up: +0 c1.b +1 c1.w +2 c2.b +3 c2.w +4 gn1.b +5 gn1.g +6 gn2.b +7 gn2.g
    //     +8 res.b +9 res.w +10 up.b +11 up.w
    auto up_block = [&](int base, int ci, int co, int H) {
        conv(B0, IN(base + 9), IN(base + 8), B1, ci, H, co, 1, 1, 0, 0);
        gn(B0, IN(base + 5), IN(base + 4), B2, ci, H * H, 1);
        conv(B2, IN(base + 1), IN(base + 0), B0, ci, H, co, 3, 1, 1, 0);
        gn(B0, IN(base + 7), IN(base + 6), B2, co, H * H, 1);
        conv(B2, IN(base + 3), IN(base + 2), B0, co, H, co, 3, 1, 1, 0);
        add(B0, B1, B2, (long)8 * co * H * H, 1);
        convt(B2, IN(base + 11), IN(base + 10), B0, co, H);                    // -> (8,co,2H,2H)
    };
    // mid: +0 agn.b +1 agn.g +2 in_b +3 in_w +4 out_b +5 out_w +6 c1.b +7 c1.w
    //      +8 c2.b +9 c2.w +10 gn1.b +11 gn1.g +12 gn2.b +13 gn2.g +14 res.b +15 res.w
    auto mid_block = [&](int base) {                                           // in B3 -> out B0
        const int H = 32, C = 256, P = 1024;
        conv(B3, IN(base + 15), IN(base + 14), B1, C, H, C, 1, 1, 0, 0);
        gn(B3, IN(base + 11), IN(base + 10), B2, C, P, 1);
        conv(B2, IN(base + 7), IN(base + 6), B0, C, H, C, 3, 1, 1, 0);
        gn(B0, IN(base + 13), IN(base + 12), B2, C, P, 1);
        conv(B2, IN(base + 9), IN(base + 8), B0, C, H, C, 3, 1, 1, 0);
        add(B0, B1, B2, (long)8 * C * P, 1);                                   // B2 = residual x
        gn(B2, IN(base + 1), IN(base + 0), B0, C, P, 0);                       // agn (no silu)
        t_fwd(B0, B1, C, P);                                                   // (8,C,P)->(8192,C)
        gemm(B1, IN(base + 3), IN(base + 2), S0, 8 * P, 3 * C, C);             // qkv
        k_attn<<<dim3(P, 2), 128, 0, stream>>>(S0, S1, P, C);                  // attention core
        gemm(S1, IN(base + 5), IN(base + 4), B1, 8 * P, C, C);                 // out proj
        t_inv(B1, B0, C, P);
        add(B0, B2, B0, (long)8 * C * P, 0);                                   // xa + res
    };

    // ======================= forward pass ==================================
    // encoder
    conv(IN(0), IN(99), IN(98), B0, 3, 256, 64, 3, 1, 1, 0);                   // enc_in
    down_block(60, 64, 128, 256, B0);                                          // enc_down[0]
    down_block(72, 128, 256, 128, B0);                                         // enc_down[1]
    down_block(84, 256, 256, 64, B3);                                          // enc_down[2] -> B3
    mid_block(100);                                                            // enc_mid -> B0
    gn(B0, IN(97), IN(96), B2, 256, 1024, 0);                                  // enc_gn
    conv(B2, IN(117), IN(116), B0, 256, 32, 3, 3, 1, 1, 0);                    // enc_out
    conv(B0, IN(121), IN(120), B2, 3, 32, 3, 1, 1, 0, 0);                      // pre_q

    // quantize
    t_fwd(B2, S0, 3, 1024);                                                    // xf rows (8192,3)
    k_zero<<<1, 32, 0, stream>>>(RED, 1);
    k_vq<<<32, 256, 0, stream>>>(S0, IN(59), S1, RED, 8192, 8192);
    t_inv(S1, OUT + ZOFF, 3, 1024);                                            // z -> output
    k_fin<<<1, 32, 0, stream>>>(RED, OUT + SOFF, 1.0f / 24576.0f);             // cb, commit

    // decoder
    conv(OUT + ZOFF, IN(119), IN(118), B0, 3, 32, 16, 1, 1, 0, 0);             // post_q
    conv(B0, IN(4), IN(3), B3, 16, 32, 256, 3, 1, 1, 0);                       // dec_in -> B3
    mid_block(5);                                                              // dec_mid -> B0
    up_block(23, 256, 256, 32);                                                // dec_up[0] -> 64
    up_block(35, 256, 128, 64);                                                // dec_up[1] -> 128
    up_block(47, 128, 64, 128);                                                // dec_up[2] -> 256
    gn(B0, IN(2), IN(1), B2, 64, 65536, 0);                                    // dec_gn
    conv(B2, IN(22), IN(21), OUT, 64, 256, 3, 3, 1, 1, 0);                     // dec_out -> d
}
